// Attention_81449759801699
// MI455X (gfx1250) — matrix operand order repaired and checked by reading, not compiled, not run on hardware
//
#include <hip/hip_runtime.h>
#include <math.h>

// ---------- types ----------
typedef __attribute__((ext_vector_type(16))) __bf16 v16bf;
typedef __attribute__((ext_vector_type(8)))  float  v8f;

struct alignas(16) U4 { unsigned int x, y, z, w; };

union Frag {
    v16bf v;
    unsigned short u[16];
    U4 q[2];
};

__device__ __forceinline__ unsigned short f32_to_bf16(float f) {
    unsigned int u = __float_as_uint(f);
    u = (u + 0x7FFFu + ((u >> 16) & 1u)) >> 16;   // round-to-nearest-even
    return (unsigned short)u;
}

__device__ __forceinline__ v8f vzero() {
    v8f z = {0.f, 0.f, 0.f, 0.f, 0.f, 0.f, 0.f, 0.f};
    return z;
}

__device__ __forceinline__ v8f wmma_bf16(const Frag& a, const Frag& b, v8f c) {
    return __builtin_amdgcn_wmma_f32_16x16x32_bf16(
        false, a.v, false, b.v, (short)0, c, false, false);
}

// ---------- problem constants ----------
#define BATCH   32
#define NTOK    1024
#define DIM     512
#define NT      400
#define TPAD    416           // NT padded to multiple of 32
#define HEADS   8
#define KDIM    64
#define DV      256
#define DH      2048          // HEADS * DV
#define HKV     2560          // HEADS * (KDIM + DV)
#define EPS     1e-5f
#define NEG_BIG (-1.0e30f)

// ---------- GEMM fragment set: 32(M) x 64(N) per wave, one 32-K chunk ----------
struct GSet { Frag a0, a1, b[4]; };

__device__ __forceinline__ void load_set(GSet& s,
                                         const unsigned short* __restrict__ a0base,
                                         const unsigned short* __restrict__ a1base,
                                         const unsigned short* __restrict__ bbase,
                                         int ldb, int kc) {
    const unsigned short* ap0 = a0base + kc;
    const unsigned short* ap1 = a1base + kc;
    s.a0.q[0] = *(const U4*)(ap0);
    s.a0.q[1] = *(const U4*)(ap0 + 16);
    s.a1.q[0] = *(const U4*)(ap1);
    s.a1.q[1] = *(const U4*)(ap1 + 16);
    const unsigned short* bp = bbase + (size_t)kc * ldb;
    int kh = (threadIdx.x & 16) >> 1;
#pragma unroll
    for (int ct = 0; ct < 4; ++ct) {
#pragma unroll
        for (int e = 0; e < 8; ++e) {
            s.b[ct].u[e]     = bp[(size_t)(kh + e) * ldb + ct * 16];
            s.b[ct].u[8 + e] = bp[(size_t)(16 + kh + e) * ldb + ct * 16];
        }
    }
}

__device__ __forceinline__ void mma_set(const GSet& s, v8f* acc0, v8f* acc1) {
#pragma unroll
    for (int ct = 0; ct < 4; ++ct) {
        acc0[ct] = wmma_bf16(s.a0, s.b[ct], acc0[ct]);
        acc1[ct] = wmma_bf16(s.a1, s.b[ct], acc1[ct]);
    }
}

// =====================================================================
// 1) fp32 -> bf16 conversion (x, q_w, proj_w)
// =====================================================================
__global__ void cvt_bf16_kernel(const float* __restrict__ src,
                                unsigned short* __restrict__ dst, int n) {
    int i = blockIdx.x * blockDim.x + threadIdx.x;
    if (i < n) dst[i] = f32_to_bf16(src[i]);
}

// =====================================================================
// 2) kv = linear_bn(text @ kv_w) -> kT (h,64,TPAD) bf16, v (h,TPAD,256) bf16
// =====================================================================
__global__ void kv_kernel(const float* __restrict__ text,
                          const float* __restrict__ kv_w,
                          const float* __restrict__ gamma,
                          const float* __restrict__ beta,
                          const float* __restrict__ mean,
                          const float* __restrict__ var,
                          unsigned short* __restrict__ kT,
                          unsigned short* __restrict__ vmat) {
    int id = blockIdx.x * blockDim.x + threadIdx.x;
    if (id >= NT * HKV) return;
    int t = id / HKV, c = id % HKV;
    const float* tr = text + t * DIM;
    float acc = 0.f;
    for (int d = 0; d < DIM; ++d) acc += tr[d] * kv_w[d * HKV + c];
    float y = (acc - mean[c]) * (gamma[c] * rsqrtf(var[c] + EPS)) + beta[c];
    int h = c / (KDIM + DV), r = c % (KDIM + DV);
    if (r < KDIM) kT[(h * KDIM + r) * TPAD + t] = f32_to_bf16(y);
    else          vmat[(h * TPAD + t) * DV + (r - KDIM)] = f32_to_bf16(y);
}

// zero pad region t in [NT, TPAD)
__global__ void pad_kernel(unsigned short* __restrict__ kT,
                           unsigned short* __restrict__ vmat) {
    int id = blockIdx.x * blockDim.x + threadIdx.x;
    const int nk = HEADS * KDIM * (TPAD - NT);          // 8192
    const int nv = HEADS * (TPAD - NT) * DV;            // 32768
    if (id < nk) {
        int t = NT + id % (TPAD - NT);
        int hk = id / (TPAD - NT);                      // h*64+kd
        kT[hk * TPAD + t] = 0;
    } else if (id < nk + nv) {
        int j = id - nk;
        int d = j % DV;
        int t = NT + (j / DV) % (TPAD - NT);
        int h = j / (DV * (TPAD - NT));
        vmat[(h * TPAD + t) * DV + d] = 0;
    }
}

// =====================================================================
// 3) bias_full[h][n][t], batch-invariant; pad cols get -1e30
// =====================================================================
__global__ void bias_kernel(const float* __restrict__ ab,
                            const int* __restrict__ Hp,
                            const int* __restrict__ Wp,
                            float* __restrict__ bias) {
    int id = blockIdx.x * blockDim.x + threadIdx.x;
    if (id >= HEADS * NTOK * TPAD) return;
    int t = id % TPAD;
    int n = (id / TPAD) % NTOK;
    int h = id / (TPAD * NTOK);
    float out;
    if (t >= NT) {
        out = NEG_BIG;
    } else {
        int W = Wp[0]; (void)Hp;
        int i = n / W, j = n - i * W;
        int a = t / 100, bcol = t - a * 100;
        int di = i - a;  di = di < 0 ? -di : di;
        int dj = j - bcol; dj = dj < 0 ? -dj : dj;
        out = ab[h * 10000 + di * 100 + dj];
    }
    bias[id] = out;
}

// =====================================================================
// 4) q = (x @ q_w -> BN -> *scale) stored bf16 as (b,h,n,64)
//    one wave: 32(M) x 64(N) tile, double-buffered K pipeline
// =====================================================================
__global__ __launch_bounds__(256) void qproj_kernel(
    const unsigned short* __restrict__ xb,
    const unsigned short* __restrict__ wb,
    const float* __restrict__ gamma, const float* __restrict__ beta,
    const float* __restrict__ mean,  const float* __restrict__ var,
    unsigned short* __restrict__ qout) {
    int lane = threadIdx.x & 31;
    int g = blockIdx.x * (blockDim.x >> 5) + (threadIdx.x >> 5);
    int rowTile = g >> 3;             // 0..1023
    int colGroup = g & 7;             // 64 cols each
    int row0 = rowTile << 5;          // 32 rows
    int koff = (lane >> 4) << 3;

    const unsigned short* a0base = xb + (size_t)(row0 + (lane & 15)) * DIM + koff;
    const unsigned short* a1base = a0base + (size_t)16 * DIM;
    const unsigned short* bbase  = wb + (lane & 15) + (colGroup << 6);

    v8f acc0[4], acc1[4];
#pragma unroll
    for (int i = 0; i < 4; ++i) { acc0[i] = vzero(); acc1[i] = vzero(); }

    GSet S, T;
    load_set(S, a0base, a1base, bbase, 512, 0);
#pragma unroll 1
    for (int kc = 0; kc < DIM; kc += 64) {
        load_set(T, a0base, a1base, bbase, 512, kc + 32);
        mma_set(S, acc0, acc1);
        if (kc + 64 < DIM) load_set(S, a0base, a1base, bbase, 512, kc + 64);
        mma_set(T, acc0, acc1);
    }

    int N = lane & 15, mb = (lane >> 4) << 3;
#pragma unroll
    for (int ct = 0; ct < 4; ++ct) {
        int c = (colGroup << 6) + ct * 16 + N;
        float gb = gamma[c] * rsqrtf(var[c] + EPS);
        float mn = mean[c], bt = beta[c];
        int h = c >> 6, kd = c & 63;
#pragma unroll
        for (int vv = 0; vv < 8; ++vv) {
            float y0 = ((acc0[ct][vv] - mn) * gb + bt) * 0.125f;  // scale=KDIM^-0.5
            float y1 = ((acc1[ct][vv] - mn) * gb + bt) * 0.125f;
            int n0 = row0 + mb + vv;
            int n1 = n0 + 16;
            int b0i = n0 >> 10, nn0 = n0 & 1023;
            int b1i = n1 >> 10, nn1 = n1 & 1023;
            qout[((((b0i << 3) + h) << 10) + nn0) * KDIM + kd] = f32_to_bf16(y0);
            qout[((((b1i << 3) + h) << 10) + nn1) * KDIM + kd] = f32_to_bf16(y1);
        }
    }
}

// =====================================================================
// 5) fused attention: scores(WMMA)+bias -> softmax(LDS) -> attn@v(WMMA)
//    -> 1/sum + GELU -> act bf16 (b, n, h*256+d).  2 waves / 64 threads.
// =====================================================================
__global__ __launch_bounds__(64) void attn_kernel(
    const unsigned short* __restrict__ qb,
    const unsigned short* __restrict__ kT,
    const unsigned short* __restrict__ vb,
    const float* __restrict__ bias,
    unsigned short* __restrict__ act) {
    __shared__ __align__(16) float s_sc[2][16][TPAD];
    __shared__ float s_inv[2][16];

    int lane = threadIdx.x & 31;
    int warp = threadIdx.x >> 5;
    int g = blockIdx.x * 2 + warp;        // 0..16383
    int b = g >> 9;
    int h = (g >> 6) & 7;
    int n0 = (g & 63) << 4;

    int rA = lane & 15;
    int koff = (lane >> 4) << 3;
    int N = lane & 15, mb = (lane >> 4) << 3;

    // q A-fragments (16 x 64 = two K-chunks of 32)
    const unsigned short* qp = qb + ((((b << 3) + h) << 10) + n0 + rA) * KDIM;
    Frag a0, a1;
    a0.q[0] = *(const U4*)(qp + koff);
    a0.q[1] = *(const U4*)(qp + koff + 16);
    a1.q[0] = *(const U4*)(qp + 32 + koff);
    a1.q[1] = *(const U4*)(qp + 32 + koff + 16);

    const unsigned short* kbase = kT + h * KDIM * TPAD;
    const float* brow = bias + (h * NTOK + n0) * TPAD;

    auto loadB = [&](Frag& b0, Frag& b1, int tt) {
        const unsigned short* kp = kbase + tt * 16 + N;
#pragma unroll
        for (int e = 0; e < 8; ++e) {
            b0.u[e]     = kp[(koff + e) * TPAD];
            b0.u[8 + e] = kp[(16 + koff + e) * TPAD];
            b1.u[e]     = kp[(32 + koff + e) * TPAD];
            b1.u[8 + e] = kp[(48 + koff + e) * TPAD];
        }
    };
    auto storeTile = [&](const v8f& acc, int tt) {
#pragma unroll
        for (int vv = 0; vv < 8; ++vv) {
            int M = mb + vv;
            s_sc[warp][M][tt * 16 + N] = acc[vv] + brow[M * TPAD + tt * 16 + N];
        }
    };

    // ---- scores: 26 tiles of 16 cols, double-buffered ----
    Frag p0, p1, q0, q1;
    loadB(p0, p1, 0);
#pragma unroll 1
    for (int tt = 0; tt < 26; tt += 2) {
        loadB(q0, q1, tt + 1);
        {
            v8f acc = vzero();
            acc = wmma_bf16(a0, p0, acc);
            acc = wmma_bf16(a1, p1, acc);
            storeTile(acc, tt);
        }
        if (tt + 2 < 26) loadB(p0, p1, tt + 2);
        {
            v8f acc = vzero();
            acc = wmma_bf16(a0, q0, acc);
            acc = wmma_bf16(a1, q1, acc);
            storeTile(acc, tt + 1);
        }
    }
    __syncthreads();

    // ---- softmax: all 32 lanes; (row = lane&15, half = lane>>4) ----
    {
        int r = lane & 15;
        int half = lane >> 4;
        float* row = s_sc[warp][r];
        int t0 = half * (TPAD / 2);
        float m = -3.0e38f;
        for (int t = t0; t < t0 + TPAD / 2; ++t) m = fmaxf(m, row[t]);
        m = fmaxf(m, __shfl_xor(m, 16, 32));
        float s = 0.f;
        for (int t = t0; t < t0 + TPAD / 2; ++t) {
            float e = __expf(row[t] - m);
            row[t] = e;
            s += e;
        }
        s += __shfl_xor(s, 16, 32);
        if (half == 0) s_inv[warp][r] = 1.0f / s;
    }
    __syncthreads();

    // ---- attn @ v : 13 K-chunks of 32, 16 d-tiles, double-buffered B ----
    v8f o[16];
#pragma unroll
    for (int i = 0; i < 16; ++i) o[i] = vzero();

#pragma unroll 1
    for (int tc = 0; tc < 13; ++tc) {
        Frag a;
        const float* arow = &s_sc[warp][rA][tc * 32 + koff];
#pragma unroll
        for (int e = 0; e < 8; ++e) a.u[e] = f32_to_bf16(arow[e]);
#pragma unroll
        for (int e = 0; e < 8; ++e) a.u[8 + e] = f32_to_bf16(arow[16 + e]);

        const unsigned short* vrow = vb + (h * TPAD + tc * 32) * DV + N;
        Frag e0, e1;
#pragma unroll
        for (int e = 0; e < 8; ++e) { e0.u[e] = vrow[(koff + e) * DV]; e0.u[8 + e] = vrow[(16 + koff + e) * DV]; }
#pragma unroll
        for (int dt = 0; dt < 16; dt += 2) {
#pragma unroll
            for (int e = 0; e < 8; ++e) { e1.u[e] = vrow[(koff + e) * DV + (dt + 1) * 16]; e1.u[8 + e] = vrow[(16 + koff + e) * DV + (dt + 1) * 16]; }
            o[dt] = wmma_bf16(a, e0, o[dt]);
            if (dt + 2 < 16) {
#pragma unroll
                for (int e = 0; e < 8; ++e) { e0.u[e] = vrow[(koff + e) * DV + (dt + 2) * 16]; e0.u[8 + e] = vrow[(16 + koff + e) * DV + (dt + 2) * 16]; }
            }
            o[dt + 1] = wmma_bf16(a, e1, o[dt + 1]);
        }
    }

    // ---- epilogue: 1/sum, exact GELU, store act bf16 ----
    unsigned short* abase = act + (((size_t)(b << 10) + n0) * DH) + (h << 8);
#pragma unroll
    for (int dt = 0; dt < 16; ++dt) {
#pragma unroll
        for (int vv = 0; vv < 8; ++vv) {
            int M = mb + vv;
            float val = o[dt][vv] * s_inv[warp][M];
            float ge = 0.5f * val * (1.0f + erff(val * 0.70710678118f));
            abase[(size_t)M * DH + dt * 16 + N] = f32_to_bf16(ge);
        }
    }
}

// =====================================================================
// 6) out = linear_bn(act @ proj_w) -> fp32
//    one wave: 32(M) x 64(N) tile, double-buffered K pipeline
// =====================================================================
__global__ __launch_bounds__(256) void proj_kernel(
    const unsigned short* __restrict__ actb,
    const unsigned short* __restrict__ wb,
    const float* __restrict__ gamma, const float* __restrict__ beta,
    const float* __restrict__ mean,  const float* __restrict__ var,
    float* __restrict__ out) {
    int lane = threadIdx.x & 31;
    int g = blockIdx.x * (blockDim.x >> 5) + (threadIdx.x >> 5);
    int rowTile = g >> 3;             // 0..1023
    int colGroup = g & 7;
    int row0 = rowTile << 5;
    int koff = (lane >> 4) << 3;

    const unsigned short* a0base = actb + (size_t)(row0 + (lane & 15)) * DH + koff;
    const unsigned short* a1base = a0base + (size_t)16 * DH;
    const unsigned short* bbase  = wb + (lane & 15) + (colGroup << 6);

    v8f acc0[4], acc1[4];
#pragma unroll
    for (int i = 0; i < 4; ++i) { acc0[i] = vzero(); acc1[i] = vzero(); }

    GSet S, T;
    load_set(S, a0base, a1base, bbase, DIM, 0);
#pragma unroll 1
    for (int kc = 0; kc < DH; kc += 64) {
        load_set(T, a0base, a1base, bbase, DIM, kc + 32);
        mma_set(S, acc0, acc1);
        if (kc + 64 < DH) load_set(S, a0base, a1base, bbase, DIM, kc + 64);
        mma_set(T, acc0, acc1);
    }

    int N = lane & 15, mb = (lane >> 4) << 3;
#pragma unroll
    for (int ct = 0; ct < 4; ++ct) {
        int c = (colGroup << 6) + ct * 16 + N;
        float gb = gamma[c] * rsqrtf(var[c] + EPS);
        float mn = mean[c], bt = beta[c];
#pragma unroll
        for (int vv = 0; vv < 8; ++vv) {
            int n0 = row0 + mb + vv;
            out[(size_t)n0 * DIM + c]        = (acc0[ct][vv] - mn) * gb + bt;
            out[(size_t)(n0 + 16) * DIM + c] = (acc1[ct][vv] - mn) * gb + bt;
        }
    }
}

// =====================================================================
// launch
// =====================================================================
extern "C" void kernel_launch(void* const* d_in, const int* in_sizes, int n_in,
                              void* d_out, int out_size, void* d_ws, size_t ws_size,
                              hipStream_t stream) {
    (void)in_sizes; (void)n_in; (void)out_size; (void)ws_size;
    const float* x        = (const float*)d_in[0];
    const float* text     = (const float*)d_in[1];
    const float* q_w      = (const float*)d_in[2];
    const float* q_gamma  = (const float*)d_in[3];
    const float* q_beta   = (const float*)d_in[4];
    const float* q_mean   = (const float*)d_in[5];
    const float* q_var    = (const float*)d_in[6];
    const float* kv_w     = (const float*)d_in[7];
    const float* kv_gamma = (const float*)d_in[8];
    const float* kv_beta  = (const float*)d_in[9];
    const float* kv_mean  = (const float*)d_in[10];
    const float* kv_var   = (const float*)d_in[11];
    const float* proj_w   = (const float*)d_in[12];
    const float* p_gamma  = (const float*)d_in[13];
    const float* p_beta   = (const float*)d_in[14];
    const float* p_mean   = (const float*)d_in[15];
    const float* p_var    = (const float*)d_in[16];
    const float* ab       = (const float*)d_in[17];
    const int*   Hp       = (const int*)d_in[18];
    const int*   Wp       = (const int*)d_in[19];
    float* out = (float*)d_out;

    char* ws = (char*)d_ws;
    size_t off = 0;
    auto alloc = [&](size_t bytes) -> void* {
        void* p = ws + off;
        off = (off + bytes + 255) & ~(size_t)255;
        return p;
    };

    unsigned short* xb    = (unsigned short*)alloc((size_t)BATCH * NTOK * DIM * 2);
    unsigned short* qwb   = (unsigned short*)alloc((size_t)DIM * 512 * 2);
    unsigned short* pwb   = (unsigned short*)alloc((size_t)DH * DIM * 2);
    unsigned short* qb    = (unsigned short*)alloc((size_t)BATCH * HEADS * NTOK * KDIM * 2);
    unsigned short* kTb   = (unsigned short*)alloc((size_t)HEADS * KDIM * TPAD * 2);
    unsigned short* vbuf  = (unsigned short*)alloc((size_t)HEADS * TPAD * DV * 2);
    float*          biasf = (float*)alloc((size_t)HEADS * NTOK * TPAD * 4);
    unsigned short* actb  = (unsigned short*)alloc((size_t)BATCH * NTOK * DH * 2);

    // 1) conversions
    {
        int n = BATCH * NTOK * DIM;
        cvt_bf16_kernel<<<(n + 255) / 256, 256, 0, stream>>>(x, xb, n);
        n = DIM * 512;
        cvt_bf16_kernel<<<(n + 255) / 256, 256, 0, stream>>>(q_w, qwb, n);
        n = DH * DIM;
        cvt_bf16_kernel<<<(n + 255) / 256, 256, 0, stream>>>(proj_w, pwb, n);
    }
    // 2) kv + padding
    pad_kernel<<<(HEADS * KDIM * 16 + HEADS * 16 * DV + 255) / 256, 256, 0, stream>>>(kTb, vbuf);
    kv_kernel<<<(NT * HKV + 255) / 256, 256, 0, stream>>>(
        text, kv_w, kv_gamma, kv_beta, kv_mean, kv_var, kTb, vbuf);
    // 3) bias table
    bias_kernel<<<(HEADS * NTOK * TPAD + 255) / 256, 256, 0, stream>>>(ab, Hp, Wp, biasf);
    // 4) q projection (8192 waves, 32x64 tile each)
    qproj_kernel<<<1024, 256, 0, stream>>>(xb, qwb, q_gamma, q_beta, q_mean, q_var, qb);
    // 5) fused attention (16384 waves, 2 waves/block)
    attn_kernel<<<8192, 64, 0, stream>>>(qb, kTb, vbuf, biasf, actb);
    // 6) output projection (8192 waves, 32x64 tile each)
    proj_kernel<<<1024, 256, 0, stream>>>(actb, pwb, p_gamma, p_beta, p_mean, p_var, out);
}
